// MultiHeadAttention_48326972014847
// MI455X (gfx1250) — matrix operand order repaired and checked by reading, not compiled, not run on hardware
//
#include <hip/hip_runtime.h>
#include <hip/hip_bf16.h>
#include <math.h>

// ---------------- problem constants ----------------
#define BB   8
#define HH   4
#define SS   1024
#define DD   512
#define HD   128
#define NBH  (BB * HH)                         // 32 (b,h) pairs
#define NSC  ((size_t)NBH * SS * SS)           // 33,554,432 scores
#define NBIN (1u << 22)                        // histogram bins (top 22 float bits)

typedef __attribute__((ext_vector_type(16))) __bf16 v16bf;
typedef __attribute__((ext_vector_type(8)))  __bf16 v8bf;
typedef __attribute__((ext_vector_type(8)))  float  v8f;
typedef __attribute__((ext_vector_type(4)))  float  v4f;

union frag_u { v16bf bf; unsigned short us[16]; };

// ---------------- helpers ----------------
__device__ __forceinline__ unsigned short f2bf_raw(float f) {
    union { float f; unsigned u; } v; v.f = f;
    unsigned r = v.u + 0x7FFFu + ((v.u >> 16) & 1u);   // round-to-nearest-even
    return (unsigned short)(r >> 16);
}

__device__ __forceinline__ v16bf cat8(v8bf lo, v8bf hi) {
    return __builtin_shufflevector(lo, hi, 0,1,2,3,4,5,6,7,8,9,10,11,12,13,14,15);
}

// A-matrix fragment (16x32 bf16), ISA 7.12.2 layout:
// lanes 0-15 hold row M=lane, K = k0+{0..7, 16..23}; lanes 16-31: K = k0+{8..15, 24..31}
__device__ __forceinline__ v16bf load_fragA(const unsigned short* mat, int ld,
                                            int r0, int k0, int lane) {
    int half = lane >> 4;
    const unsigned short* p = mat + (size_t)(r0 + (lane & 15)) * ld + k0 + half * 8;
    v8bf lo = *(const v8bf*)(p);
    v8bf hi = *(const v8bf*)(p + 16);
    return cat8(lo, hi);
}

// B-matrix fragment (32x16 bf16): sources are row-major with the N-dimension as
// rows, lane l loads row n0+(l&15); lanes 0-15 K=k0+{0..7,16..23}, lanes 16-31 K=k0+{8..15,24..31}.
__device__ __forceinline__ v16bf load_fragB(const unsigned short* mat, int ld,
                                            int n0, int k0, int lane) {
    int half = lane >> 4;
    const unsigned short* p = mat + (size_t)(n0 + (lane & 15)) * ld + k0 + half * 8;
    v8bf lo = *(const v8bf*)(p);
    v8bf hi = *(const v8bf*)(p + 16);
    return cat8(lo, hi);
}

__device__ __forceinline__ v8f wmma_bf16(v16bf a, v16bf b, v8f c) {
    return __builtin_amdgcn_wmma_f32_16x16x32_bf16(false, a, false, b,
                                                   (short)0, c, false, false);
}

// rank transform: deterministic mid-rank of the element's 22-bit |x| bin
__device__ __forceinline__ float rank_xform(float x, const unsigned* __restrict__ hist,
                                            const unsigned* __restrict__ excl) {
    unsigned bits = __float_as_uint(x) & 0x7FFFFFFFu;
    unsigned bin  = (NBIN - 1u) - (bits >> 9);
    float rank = (float)excl[bin] + 0.5f * ((float)hist[bin] + 1.0f);
    float sgn  = (x > 0.f) ? 1.f : ((x < 0.f) ? -1.f : 0.f);
    return -logf(rank * (1.0f / (float)NSC)) * sgn;
}

// A fragment loaded from fp32 scores with the rank transform fused in-register
__device__ __forceinline__ v16bf load_fragA_xf(const float* __restrict__ mat, int ld,
                                               int r0, int k0, int lane,
                                               const unsigned* __restrict__ hist,
                                               const unsigned* __restrict__ excl) {
    int half = lane >> 4;
    const float* p = mat + (size_t)(r0 + (lane & 15)) * ld + k0 + half * 8;
    v4f x0 = *(const v4f*)(p);
    v4f x1 = *(const v4f*)(p + 4);
    v4f x2 = *(const v4f*)(p + 16);
    v4f x3 = *(const v4f*)(p + 20);
    frag_u f;
#pragma unroll
    for (int i = 0; i < 4; ++i) {
        f.us[i]      = f2bf_raw(rank_xform(x0[i], hist, excl));
        f.us[4 + i]  = f2bf_raw(rank_xform(x1[i], hist, excl));
        f.us[8 + i]  = f2bf_raw(rank_xform(x2[i], hist, excl));
        f.us[12 + i] = f2bf_raw(rank_xform(x3[i], hist, excl));
    }
    return f.bf;
}

// ---------------- kernel 0: clear histogram ----------------
__global__ void clear_u32_kernel(unsigned* p, unsigned n) {
    unsigned i = blockIdx.x * blockDim.x + threadIdx.x;
    if (i < n) p[i] = 0u;
}

// ---------------- kernel 1: norms + normalized bf16 Q/K + transposed bf16 V ----
__global__ __launch_bounds__(128)
void prep_kernel(const float* __restrict__ q, const float* __restrict__ k,
                 const float* __restrict__ v,
                 unsigned short* __restrict__ Qn, unsigned short* __restrict__ Kn,
                 unsigned short* __restrict__ Vt) {
    const int ROWS = NBH * SS;                 // 32768 = 2^15
    int rid   = blockIdx.x;
    int which = rid >> 15;                     // 0=q, 1=k, 2=v
    int bhs   = rid & (ROWS - 1);
    int b  = bhs / (HH * SS);
    int hs = bhs % (HH * SS);
    int h  = hs / SS;
    int s  = hs % SS;
    int d  = threadIdx.x;

    const float* src = (which == 0) ? q : (which == 1) ? k : v;
    float x = src[((size_t)b * SS + s) * DD + h * HD + d];

    if (which < 2) {
        __shared__ float red[128];
        red[d] = x * x;
        __syncthreads();
        for (int t = 64; t > 0; t >>= 1) {
            if (d < t) red[d] += red[d + t];
            __syncthreads();
        }
        float inv = rsqrtf(red[0]);            // fold 1/|row| into operand
        unsigned short bv = f2bf_raw(x * inv);
        size_t base = ((size_t)(b * HH + h) * SS + s) * HD + d;
        if (which == 0) Qn[base] = bv; else Kn[base] = bv;
    } else {
        // Vt[bh][d][s] so GEMM2's B operand loads row-major like GEMM1's
        Vt[((size_t)(b * HH + h) * HD + d) * SS + s] = f2bf_raw(x);
    }
}

// ---------------- kernel 2: GEMM1 = cosine scores + histogram ----------------
__global__ __launch_bounds__(256)
void gemm1_kernel(const unsigned short* __restrict__ Qn,
                  const unsigned short* __restrict__ Kn,
                  float* __restrict__ scores, unsigned* __restrict__ hist) {
    int wave = threadIdx.x >> 5;
    int lane = threadIdx.x & 31;
    int half = lane >> 4;
    int bh   = blockIdx.x >> 7;                       // 128 blocks per (b,h)
    int t    = ((blockIdx.x & 127) << 3) | wave;      // wave tile 0..1023
    int q0   = (t >> 5) * 32;
    int c0   = (t & 31) * 32;

    const unsigned short* Q = Qn + (size_t)bh * SS * HD;
    const unsigned short* K = Kn + (size_t)bh * SS * HD;

    v8f acc[2][2] = {};
    for (int kk = 0; kk < HD; kk += 32) {
        v16bf a0 = load_fragA(Q, HD, q0,      kk, lane);
        v16bf a1 = load_fragA(Q, HD, q0 + 16, kk, lane);
        v16bf b0 = load_fragB(K, HD, c0,      kk, lane);
        v16bf b1 = load_fragB(K, HD, c0 + 16, kk, lane);
        acc[0][0] = wmma_bf16(a0, b0, acc[0][0]);
        acc[0][1] = wmma_bf16(a0, b1, acc[0][1]);
        acc[1][0] = wmma_bf16(a1, b0, acc[1][0]);
        acc[1][1] = wmma_bf16(a1, b1, acc[1][1]);
    }

    float* sc = scores + (size_t)bh * SS * SS;
    for (int i = 0; i < 2; ++i)
        for (int j = 0; j < 2; ++j)
            for (int vv = 0; vv < 8; ++vv) {
                int r = q0 + 16 * i + vv + 8 * half;
                int c = c0 + 16 * j + (lane & 15);
                float x = acc[i][j][vv];
                sc[(size_t)r * SS + c] = x;
                unsigned bits = __float_as_uint(x) & 0x7FFFFFFFu;
                unsigned bin  = (NBIN - 1u) - (bits >> 9);   // ascending bin == descending |x|
                atomicAdd(&hist[bin], 1u);
            }
}

// ---------------- kernels 3-5: hierarchical exclusive scan over 2^22 bins ----
#define SCAN_CHUNK  4096
#define SCAN_BLOCKS (NBIN / SCAN_CHUNK)        // 1024

__global__ __launch_bounds__(256)
void scan_part_kernel(const unsigned* __restrict__ hist, unsigned* __restrict__ part) {
    __shared__ unsigned red[256];
    unsigned base = blockIdx.x * SCAN_CHUNK;
    unsigned sum = 0;
    for (int i = threadIdx.x; i < SCAN_CHUNK; i += 256) sum += hist[base + i];
    red[threadIdx.x] = sum;
    __syncthreads();
    for (int s = 128; s > 0; s >>= 1) {
        if ((int)threadIdx.x < s) red[threadIdx.x] += red[threadIdx.x + s];
        __syncthreads();
    }
    if (threadIdx.x == 0) part[blockIdx.x] = red[0];
}

__global__ void scan_tops_kernel(unsigned* part) {
    if (threadIdx.x == 0) {
        unsigned run = 0;
        for (int i = 0; i < SCAN_BLOCKS; ++i) { unsigned v = part[i]; part[i] = run; run += v; }
    }
}

__global__ __launch_bounds__(256)
void scan_final_kernel(const unsigned* __restrict__ hist, const unsigned* __restrict__ part,
                       unsigned* __restrict__ excl) {
    __shared__ unsigned tsum[256];
    __shared__ unsigned tex[256];
    unsigned base = blockIdx.x * SCAN_CHUNK;
    unsigned off  = threadIdx.x * 16;          // 16 consecutive bins per thread
    unsigned local[16];
    unsigned s = 0;
    for (int i = 0; i < 16; ++i) { local[i] = s; s += hist[base + off + i]; }
    tsum[threadIdx.x] = s;
    __syncthreads();
    if (threadIdx.x == 0) {
        unsigned run = 0;
        for (int i = 0; i < 256; ++i) { tex[i] = run; run += tsum[i]; }
    }
    __syncthreads();
    unsigned add = part[blockIdx.x] + tex[threadIdx.x];
    for (int i = 0; i < 16; ++i) excl[base + off + i] = add + local[i];
}

// ---------------- kernel 6: GEMM2 with fused rank transform ----------------
// out = (-log(rank/n)*sign(score)) @ V.  Each wave owns a 32x128 output stripe
// (2x8 WMMA accumulators), so every fp32 score element is loaded exactly once
// and the rank transform happens in-register (no intermediate buffer).
__global__ __launch_bounds__(256)
void gemm2_fused_kernel(const float* __restrict__ scores,
                        const unsigned short* __restrict__ Vt,
                        const unsigned* __restrict__ hist,
                        const unsigned* __restrict__ excl,
                        float* __restrict__ out) {
    int wave = threadIdx.x >> 5;
    int lane = threadIdx.x & 31;
    int half = lane >> 4;
    int bh   = blockIdx.x >> 2;                   // 4 blocks (32 waves) per (b,h)
    int t    = ((blockIdx.x & 3) << 3) | wave;    // row tile 0..31
    int q0   = t * 32;

    const float*          A  = scores + (size_t)bh * SS * SS;
    const unsigned short* Bm = Vt + (size_t)bh * HD * SS;

    v8f acc[2][8] = {};
    for (int kk = 0; kk < SS; kk += 32) {
        v16bf a0 = load_fragA_xf(A, SS, q0,      kk, lane, hist, excl);
        v16bf a1 = load_fragA_xf(A, SS, q0 + 16, kk, lane, hist, excl);
#pragma unroll
        for (int j = 0; j < 8; ++j) {
            v16bf b = load_fragB(Bm, SS, j * 16, kk, lane);
            acc[0][j] = wmma_bf16(a0, b, acc[0][j]);
            acc[1][j] = wmma_bf16(a1, b, acc[1][j]);
        }
    }

    int b = bh / HH, h = bh % HH;
#pragma unroll
    for (int i = 0; i < 2; ++i)
#pragma unroll
        for (int j = 0; j < 8; ++j)
#pragma unroll
            for (int vv = 0; vv < 8; ++vv) {
                int r = q0 + 16 * i + vv + 8 * half;
                int c = 16 * j + (lane & 15);
                out[((size_t)b * SS + r) * DD + h * HD + c] = acc[i][j][vv];
            }
}

// ---------------- host launcher ----------------
extern "C" void kernel_launch(void* const* d_in, const int* in_sizes, int n_in,
                              void* d_out, int out_size, void* d_ws, size_t ws_size,
                              hipStream_t stream) {
    (void)in_sizes; (void)n_in; (void)out_size; (void)ws_size;
    const float* q = (const float*)d_in[0];
    const float* k = (const float*)d_in[1];
    const float* v = (const float*)d_in[2];
    float* out = (float*)d_out;

    char* ws = (char*)d_ws;
    size_t off = 0;
    auto take = [&](size_t bytes) -> char* {
        char* p = ws + off;
        off = (off + bytes + 255) & ~(size_t)255;
        return p;
    };
    unsigned short* Qn     = (unsigned short*)take((size_t)NBH * SS * HD * 2);  //   8 MB
    unsigned short* Kn     = (unsigned short*)take((size_t)NBH * SS * HD * 2);  //   8 MB
    unsigned short* Vt     = (unsigned short*)take((size_t)NBH * HD * SS * 2);  //   8 MB
    float*          scores = (float*)take(NSC * 4);                             // 128 MB
    unsigned*       hist   = (unsigned*)take((size_t)NBIN * 4);                 //  16 MB
    unsigned*       excl   = (unsigned*)take((size_t)NBIN * 4);                 //  16 MB
    unsigned*       part   = (unsigned*)take((size_t)SCAN_BLOCKS * 4);

    clear_u32_kernel<<<NBIN / 256, 256, 0, stream>>>(hist, NBIN);
    prep_kernel<<<3 * NBH * SS, 128, 0, stream>>>(q, k, v, Qn, Kn, Vt);
    gemm1_kernel<<<NBH * 128, 256, 0, stream>>>(Qn, Kn, scores, hist);
    scan_part_kernel<<<SCAN_BLOCKS, 256, 0, stream>>>(hist, part);
    scan_tops_kernel<<<1, 32, 0, stream>>>(part);
    scan_final_kernel<<<SCAN_BLOCKS, 256, 0, stream>>>(hist, part, excl);
    gemm2_fused_kernel<<<NBH * 4, 256, 0, stream>>>(scores, Vt, hist, excl, out);
}
